// Spe_SSM_32469952757783
// MI455X (gfx1250) — matrix operand order repaired and checked by reading, not compiled, not run on hardware
//
#include <hip/hip_runtime.h>
#include <hip/hip_bf16.h>

// ---------------------------------------------------------------------------
// MI455X (gfx1250) vision-SSM block.
// Bandwidth-bound (~350MB traffic vs ~35 GFLOP): GEMMs use
// V_WMMA_F32_16X16X32_F16 with f32 accumulate; LayerNorms feeding a GEMM are
// folded into the A-operand load. Weights are pre-transposed to [N][K] f16 so
// both WMMA operands are K-contiguous per lane (b128 LDS loads only), and
// weight tiles are staged global->LDS with CDNA5 async-to-LDS (ASYNCcnt).
// ---------------------------------------------------------------------------

typedef __attribute__((ext_vector_type(16))) _Float16 v16h;
typedef __attribute__((ext_vector_type(8)))  _Float16 v8h;
typedef __attribute__((ext_vector_type(4)))  _Float16 v4h;
typedef __attribute__((ext_vector_type(8)))  float    v8f;

#define DEV static __device__ __forceinline__

DEV float gelu_exact(float x) { return 0.5f * x * (1.0f + erff(x * 0.70710678118f)); }
DEV float silu(float x)       { return x / (1.0f + expf(-x)); }
DEV float softplus(float x)   { return (x > 20.0f) ? x : log1pf(expf(x)); }

// CDNA5 async global->LDS copy, 16B per lane (ISA 15.18.3 op 98, ASYNCcnt).
DEV void async_ld_b128(void* lds_ptr, const void* gaddr) {
    unsigned lds = (unsigned)(unsigned long long)lds_ptr;  // LDS offset VGPR
    asm volatile("global_load_async_to_lds_b128 %0, %1, off"
                 :: "v"(lds), "v"(gaddr) : "memory");
}
DEV void wait_asynccnt0() { asm volatile("s_wait_asynccnt 0" ::: "memory"); }

// ---------------------------------------------------------------------------
// f32 [K][N] -> f16 transposed [N][K] (one-time weight prep)
// ---------------------------------------------------------------------------
__global__ void cvt_tr_kernel(const float* __restrict__ w, _Float16* __restrict__ o,
                              unsigned K, unsigned N) {
    unsigned i = blockIdx.x * 256u + threadIdx.x;          // over N*K, output-coalesced
    if (i >= K * N) return;
    unsigned k = i % K, n = i / K;
    o[i] = (_Float16)w[(size_t)k * N + n];
}

// ---------------------------------------------------------------------------
// WMMA GEMM:  C[M,N] = epilogue( lnA(A)[M,K] @ W[K,N] + bias ), W given as
// Wt[N][K] f16.  256 threads = 8 waves; block tile 64(M) x 64(N); K-step 32.
// Wave (wm,wn): 16 rows x 32 cols -> 2 WMMAs sharing one A fragment.
//   A: float4 global loads -> fused LN -> f16 -> b64 LDS stores.
//   B: async b128 global->LDS ([n][k], K-contiguous).
// Fragments use contiguous k = 16*(lane/16) + s for both A and B; all LDS reads are ds_load_b128.
// ---------------------------------------------------------------------------
template <int HAS_LN, int HAS_BIAS, int DO_GELU>
__global__ __launch_bounds__(256) void gemm_f16_kernel(
    const float* __restrict__ A, const _Float16* __restrict__ Wt,
    const float* __restrict__ bias,
    const float* __restrict__ stats,   // [M*2] (mean, rstd)
    const float* __restrict__ lnG, const float* __restrict__ lnB,
    float* __restrict__ C, int M, int N, int Kd)
{
    __shared__ _Float16 sA[64 * 32];   // [m][k]
    __shared__ _Float16 sBt[64 * 32];  // [n][k]

    const int tid  = threadIdx.x;
    const int lane = tid & 31;
    const int wave = tid >> 5;
    const int wm   = wave >> 1;                 // 4 waves along M
    const int wn   = wave & 1;                  // 2 waves along N
    const int bm   = blockIdx.y * 64;
    const int bn   = blockIdx.x * 64;

    // A staging role: row ra (0..63), k-chunk ca (0,8,16,24)
    const int ra = tid >> 2;
    const int ca = (tid & 3) * 8;

    float muA = 0.f, rsA = 0.f;
    if (HAS_LN) {
        muA = stats[(bm + ra) * 2 + 0];
        rsA = stats[(bm + ra) * 2 + 1];
    }

    v8f acc0 = {0.f, 0.f, 0.f, 0.f, 0.f, 0.f, 0.f, 0.f};
    v8f acc1 = {0.f, 0.f, 0.f, 0.f, 0.f, 0.f, 0.f, 0.f};

    #pragma unroll 2
    for (int k0 = 0; k0 < Kd; k0 += 32) {
        // ---- B tile: async global->LDS, 16B per lane, K-contiguous -------
        async_ld_b128(&sBt[ra * 32 + ca], &Wt[(size_t)(bn + ra) * Kd + (k0 + ca)]);

        // ---- A tile: float4 loads, optional fused LN, packed f16 stores --
        {
            const float* ap = A + (size_t)(bm + ra) * Kd + (k0 + ca);
            float4 a0 = *(const float4*)(ap + 0);
            float4 a1 = *(const float4*)(ap + 4);
            if (HAS_LN) {
                float4 g0 = *(const float4*)(lnG + k0 + ca);
                float4 g1 = *(const float4*)(lnG + k0 + ca + 4);
                float4 b0 = *(const float4*)(lnB + k0 + ca);
                float4 b1 = *(const float4*)(lnB + k0 + ca + 4);
                a0.x = (a0.x - muA) * rsA * g0.x + b0.x;
                a0.y = (a0.y - muA) * rsA * g0.y + b0.y;
                a0.z = (a0.z - muA) * rsA * g0.z + b0.z;
                a0.w = (a0.w - muA) * rsA * g0.w + b0.w;
                a1.x = (a1.x - muA) * rsA * g1.x + b1.x;
                a1.y = (a1.y - muA) * rsA * g1.y + b1.y;
                a1.z = (a1.z - muA) * rsA * g1.z + b1.z;
                a1.w = (a1.w - muA) * rsA * g1.w + b1.w;
            }
            v4h h0, h1;
            h0[0] = (_Float16)a0.x; h0[1] = (_Float16)a0.y;
            h0[2] = (_Float16)a0.z; h0[3] = (_Float16)a0.w;
            h1[0] = (_Float16)a1.x; h1[1] = (_Float16)a1.y;
            h1[2] = (_Float16)a1.z; h1[3] = (_Float16)a1.w;
            *(v4h*)&sA[ra * 32 + ca]     = h0;
            *(v4h*)&sA[ra * 32 + ca + 4] = h1;
        }

        wait_asynccnt0();
        __syncthreads();

        // ---- fragments: all b128 LDS loads (contiguous k = 16h + s) ----------
        // A lanes 0-15: K 0-15; lanes 16-31: K 16-31. Per lane: two
        // contiguous 8-half chunks at khi and khi+8.
        const int khi = (lane >> 4) * 16;
        const int m   = wm * 16 + (lane & 15);
        v8h a_lo = *(const v8h*)&sA[m * 32 + khi];
        v8h a_hi = *(const v8h*)&sA[m * 32 + khi + 8];
        v16h af = __builtin_shufflevector(a_lo, a_hi,
                   0,1,2,3,4,5,6,7, 8,9,10,11,12,13,14,15);
        // B lanes 0-15: K 0-15; lanes 16-31: K 16-31 -> 16 contiguous halves.
        const int kb = (lane >> 4) * 16;
        const int n0 = wn * 32 + (lane & 15);
        v8h b0lo = *(const v8h*)&sBt[n0 * 32 + kb];
        v8h b0hi = *(const v8h*)&sBt[n0 * 32 + kb + 8];
        v8h b1lo = *(const v8h*)&sBt[(n0 + 16) * 32 + kb];
        v8h b1hi = *(const v8h*)&sBt[(n0 + 16) * 32 + kb + 8];
        v16h bf0 = __builtin_shufflevector(b0lo, b0hi,
                   0,1,2,3,4,5,6,7, 8,9,10,11,12,13,14,15);
        v16h bf1 = __builtin_shufflevector(b1lo, b1hi,
                   0,1,2,3,4,5,6,7, 8,9,10,11,12,13,14,15);

        acc0 = __builtin_amdgcn_wmma_f32_16x16x32_f16(false, af, false, bf0,
                                                      (short)0, acc0, false, false);
        acc1 = __builtin_amdgcn_wmma_f32_16x16x32_f16(false, af, false, bf1,
                                                      (short)0, acc1, false, false);
        __syncthreads();
    }

    // ---- write D (f32 16x16 layout) --------------------------------------
    const int mBase = bm + wm * 16 + ((lane >> 4) ? 8 : 0);
    #pragma unroll
    for (int sub = 0; sub < 2; ++sub) {
        const v8f& acc = sub ? acc1 : acc0;
        const int nOut = bn + wn * 32 + sub * 16 + (lane & 15);
        float bv = 0.f;
        if (HAS_BIAS) bv = bias[nOut];
        #pragma unroll
        for (int r = 0; r < 8; ++r) {
            float v = acc[r] + bv;
            if (DO_GELU) v = gelu_exact(v);
            C[(size_t)(mBase + r) * N + nOut] = v;
        }
    }
}

// ---------------------------------------------------------------------------
// Wave-per-row LayerNorm (in-place safe) and row statistics
// ---------------------------------------------------------------------------
__global__ __launch_bounds__(128) void ln_kernel(
    const float* __restrict__ X, const float* __restrict__ g,
    const float* __restrict__ b, float* __restrict__ Y, int M, int C)
{
    int row  = blockIdx.x * 4 + (threadIdx.x >> 5);
    int lane = threadIdx.x & 31;
    if (row >= M) return;
    const float* x = X + (size_t)row * C;
    float s = 0.f, s2 = 0.f;
    for (int c = lane; c < C; c += 32) { float v = x[c]; s += v; s2 += v * v; }
    #pragma unroll
    for (int o = 16; o; o >>= 1) {
        s  += __shfl_xor(s,  o, 32);
        s2 += __shfl_xor(s2, o, 32);
    }
    float mu = s / C;
    float rs = rsqrtf(s2 / C - mu * mu + 1e-5f);
    float* y = Y + (size_t)row * C;
    for (int c = lane; c < C; c += 32) y[c] = (x[c] - mu) * rs * g[c] + b[c];
}

__global__ __launch_bounds__(128) void row_stats_kernel(
    const float* __restrict__ X, float* __restrict__ stats, int M, int C)
{
    int row  = blockIdx.x * 4 + (threadIdx.x >> 5);
    int lane = threadIdx.x & 31;
    if (row >= M) return;
    const float* x = X + (size_t)row * C;
    float s = 0.f, s2 = 0.f;
    for (int c = lane; c < C; c += 32) { float v = x[c]; s += v; s2 += v * v; }
    #pragma unroll
    for (int o = 16; o; o >>= 1) {
        s  += __shfl_xor(s,  o, 32);
        s2 += __shfl_xor(s2, o, 32);
    }
    if (lane == 0) {
        float mu = s / C;
        stats[row * 2 + 0] = mu;
        stats[row * 2 + 1] = rsqrtf(s2 / C - mu * mu + 1e-5f);
    }
}

// ---------------------------------------------------------------------------
// Depthwise 3x3 conv (pad 1) + SiLU.  xz: (B,H,W,128) [first 64 = xc]
// ---------------------------------------------------------------------------
__global__ void conv_silu_kernel(const float* __restrict__ xz,
                                 const float* __restrict__ cw,
                                 const float* __restrict__ cb,
                                 float* __restrict__ xh)
{
    unsigned idx = blockIdx.x * 256u + threadIdx.x;       // 4194304
    if (idx >= 16u * 64u * 64u * 64u) return;
    int c = idx & 63, w = (idx >> 6) & 63, h = (idx >> 12) & 63, b = idx >> 18;
    float acc = cb[c];
    #pragma unroll
    for (int dy = -1; dy <= 1; ++dy)
        #pragma unroll
        for (int dx = -1; dx <= 1; ++dx) {
            int hy = h + dy, wx = w + dx;
            if ((unsigned)hy < 64u && (unsigned)wx < 64u)
                acc += xz[((size_t)((b * 64 + hy) * 64 + wx)) * 128 + c]
                     * cw[c * 9 + (dy + 1) * 3 + (dx + 1)];
        }
    xh[idx] = silu(acc);
}

// x_fb[b, k*64+c, h, w] = xh[b,h,w, k ? 63-c : c]
__global__ void xfb_kernel(const float* __restrict__ xh, float* __restrict__ xfb)
{
    unsigned idx = blockIdx.x * 256u + threadIdx.x;       // 8388608
    if (idx >= 8388608u) return;
    int w = idx & 63, h = (idx >> 6) & 63, c2 = (idx >> 12) & 127, b = idx >> 19;
    int c = (c2 >> 6) ? (63 - (c2 & 63)) : (c2 & 63);
    xfb[idx] = xh[((size_t)((b * 64 + h) * 64 + w)) * 64 + c];
}

// A_br[((b*8+hb)*8+wb)*8192 + (a*8+e)*128 + c2] = xfb[b, c2, a*8+hb, e*8+wb]
__global__ void gather_branch_kernel(const float* __restrict__ xfb,
                                     float* __restrict__ Abr)
{
    unsigned idx = blockIdx.x * 256u + threadIdx.x;       // 8388608
    if (idx >= 8388608u) return;
    int c2 = idx & 127, e = (idx >> 7) & 7, a = (idx >> 10) & 7;
    int wb = (idx >> 13) & 7, hb = (idx >> 16) & 7, b = idx >> 19;
    int H = a * 8 + hb, W = e * 8 + wb;
    Abr[idx] = xfb[(((size_t)(b * 128 + c2) * 64) + H) * 64 + W];
}

// A_pt[((b*8+hc)*8+wc)*8192 + (pi*8+pj)*128 + c2] = xfb[b, c2, hc*8+pi, wc*8+pj]
__global__ void gather_patch_kernel(const float* __restrict__ xfb,
                                    float* __restrict__ Apt)
{
    unsigned idx = blockIdx.x * 256u + threadIdx.x;       // 8388608
    if (idx >= 8388608u) return;
    int c2 = idx & 127, pj = (idx >> 7) & 7, pi = (idx >> 10) & 7;
    int wc = (idx >> 13) & 7, hc = (idx >> 16) & 7, b = idx >> 19;
    int H = hc * 8 + pi, W = wc * 8 + pj;
    Apt[idx] = xfb[(((size_t)(b * 128 + c2) * 64) + H) * 64 + W];
}

// A2[((b*2+hb2)*2+wb2)*2048 + (i4*4+j4)*128 + c]
__global__ void gather_stage2_kernel(const float* __restrict__ t,
                                     float* __restrict__ A2)
{
    unsigned idx = blockIdx.x * 256u + threadIdx.x;       // 131072
    if (idx >= 131072u) return;
    int c = idx & 127, j4 = (idx >> 7) & 3, i4 = (idx >> 9) & 3;
    int wb2 = (idx >> 11) & 1, hb2 = (idx >> 12) & 1, b = idx >> 13;
    A2[idx] = t[((size_t)((b * 8 + i4 * 2 + hb2) * 8 + (j4 * 2 + wb2))) * 128 + c];
}

__global__ void posadd_kernel(float* __restrict__ tP, const float* __restrict__ pe)
{
    unsigned idx = blockIdx.x * 256u + threadIdx.x;       // 131072
    if (idx >= 131072u) return;
    int c = idx & 127, row = idx >> 7, s = row & 63;
    tP[idx] += pe[s * 128 + c];
}

// ---------------------------------------------------------------------------
// dt path: per (b, chan) row of length 64 -> (64->8->64) MLP + LN(64)
// ---------------------------------------------------------------------------
__global__ void dt_kernel(const float* __restrict__ tP,
                          const float* __restrict__ W1, const float* __restrict__ b1,
                          const float* __restrict__ W2, const float* __restrict__ b2,
                          const float* __restrict__ g,  const float* __restrict__ bb,
                          float* __restrict__ tdout)
{
    int i = blockIdx.x * 128 + threadIdx.x;               // 2048 rows
    if (i >= 2048) return;
    int b = i >> 7, ch = i & 127;
    float td[64], t8[8], o[64];
    for (int s = 0; s < 64; ++s) td[s] = tP[((size_t)(b * 64 + s)) * 128 + ch];
    for (int j = 0; j < 8; ++j) {
        float a = b1[j];
        for (int s = 0; s < 64; ++s) a += td[s] * W1[s * 8 + j];
        t8[j] = a;
    }
    float mu = 0.f;
    for (int s = 0; s < 64; ++s) {
        float a = b2[s];
        #pragma unroll
        for (int j = 0; j < 8; ++j) a += t8[j] * W2[j * 64 + s];
        o[s] = a; mu += a;
    }
    mu /= 64.f;
    float var = 0.f;
    for (int s = 0; s < 64; ++s) { float dv = o[s] - mu; var += dv * dv; }
    float rs = rsqrtf(var / 64.f + 1e-5f);
    for (int s = 0; s < 64; ++s)
        tdout[(size_t)i * 64 + s] = (o[s] - mu) * rs * g[s] + bb[s];
}

// ---------------------------------------------------------------------------
// Selective scan: thread per (b, Dd-row), 64 steps, 4 states
// ---------------------------------------------------------------------------
__global__ void scan_kernel(const float* __restrict__ tP,  const float* __restrict__ td,
                            const float* __restrict__ Bs,  const float* __restrict__ Cs,
                            const float* __restrict__ Alog, const float* __restrict__ Ds,
                            float* __restrict__ out_y)
{
    int t = blockIdx.x * 128 + threadIdx.x;               // 2048
    if (t >= 2048) return;
    int b = t >> 7, i = t & 127;
    int k = i >> 6, s = i & 63;
    float A[4], h[4] = {0.f, 0.f, 0.f, 0.f};
    #pragma unroll
    for (int n = 0; n < 4; ++n) A[n] = -expf(Alog[i * 4 + n]);
    float Dv = Ds[i];
    for (int l = 0; l < 64; ++l) {
        float u  = tP[((size_t)(b * 64 + s)) * 128 + (k * 64 + l)];
        float dl = softplus(td[((size_t)(b * 128 + k * 64 + l)) * 64 + s]);
        float y = 0.f;
        #pragma unroll
        for (int n = 0; n < 4; ++n) {
            float Bv = Bs[((size_t)(b * 4 + n)) * 128 + k * 64 + l];
            float Cv = Cs[((size_t)(b * 4 + n)) * 128 + k * 64 + l];
            h[n] = expf(dl * A[n]) * h[n] + dl * Bv * u;
            y += h[n] * Cv;
        }
        out_y[(((size_t)((b * 2 + k) * 64 + s)) * 64) + l] = y + Dv * u;
    }
}

__global__ void gate_kernel(const float* __restrict__ out_y,
                            const float* __restrict__ gW, float* __restrict__ g)
{
    int b = blockIdx.x * 16 + threadIdx.x;
    if (b >= 16) return;
    float l0 = 0.f, l1 = 0.f;
    for (int s = 0; s < 64; ++s) {
        float m1 = 0.f, m2 = 0.f;
        for (int l = 0; l < 64; ++l) {
            m1 += out_y[(((size_t)((b * 2 + 0) * 64 + s)) * 64) + l];
            m2 += out_y[(((size_t)((b * 2 + 1) * 64 + s)) * 64) + l];
        }
        m1 *= (1.f / 64.f); m2 *= (1.f / 64.f);
        l0 += m1 * gW[s * 2 + 0] + m2 * gW[(64 + s) * 2 + 0];
        l1 += m1 * gW[s * 2 + 1] + m2 * gW[(64 + s) * 2 + 1];
    }
    float mx = fmaxf(l0, l1);
    float e0 = expf(l0 - mx), e1 = expf(l1 - mx);
    g[b * 2 + 0] = e0 / (e0 + e1);
    g[b * 2 + 1] = e1 / (e0 + e1);
}

// yc[(b*64+l)*64+s] = g0*y1[b,s,l] + g1*out_y[b,1,s,63-l]
__global__ void combine_kernel(const float* __restrict__ out_y,
                               const float* __restrict__ g, float* __restrict__ yc)
{
    unsigned idx = blockIdx.x * 256u + threadIdx.x;       // 65536
    if (idx >= 65536u) return;
    int s = idx & 63, l = (idx >> 6) & 63, b = idx >> 12;
    float v1 = out_y[(((size_t)((b * 2 + 0) * 64 + s)) * 64) + l];
    float v2 = out_y[(((size_t)((b * 2 + 1) * 64 + s)) * 64) + (63 - l)];
    yc[((size_t)(b * 64 + l)) * 64 + s] = g[b * 2] * v1 + g[b * 2 + 1] * v2;
}

// yA[(b*64+s)*64 + l] = ycln[(b*64+l)*64 + s]
__global__ void rearr_y_kernel(const float* __restrict__ ycln, float* __restrict__ yA)
{
    unsigned idx = blockIdx.x * 256u + threadIdx.x;       // 65536
    if (idx >= 65536u) return;
    int l = idx & 63, s = (idx >> 6) & 63, b = idx >> 12;
    yA[idx] = ycln[((size_t)(b * 64 + l)) * 64 + s];
}

// yin = un-patched(ybig) * silu(z)
__global__ void mulz_kernel(const float* __restrict__ ybig,
                            const float* __restrict__ xz, float* __restrict__ yin)
{
    unsigned idx = blockIdx.x * 256u + threadIdx.x;       // 4194304
    if (idx >= 4194304u) return;
    int c = idx & 63, Wi = (idx >> 6) & 63, Hi = (idx >> 12) & 63, b = idx >> 18;
    int hh = Hi >> 3, pi = Hi & 7, ww = Wi >> 3, pj = Wi & 7;
    float yv = ybig[((size_t)(b * 64 + hh * 8 + ww)) * 4096 + (pi * 8 + pj) * 64 + c];
    float zv = xz[((size_t)((b * 64 + Hi) * 64 + Wi)) * 128 + 64 + c];
    yin[idx] = yv * silu(zv);
}

// ---------------------------------------------------------------------------
// Host orchestration
// ---------------------------------------------------------------------------
extern "C" void kernel_launch(void* const* d_in, const int* in_sizes, int n_in,
                              void* d_out, int out_size, void* d_ws, size_t ws_size,
                              hipStream_t stream)
{
    (void)in_sizes; (void)n_in; (void)out_size; (void)ws_size;

    const float* x       = (const float*)d_in[0];
    const float* in_W1   = (const float*)d_in[1];
    const float* in_W2   = (const float*)d_in[2];
    const float* in_ln_g = (const float*)d_in[3];
    const float* in_ln_b = (const float*)d_in[4];
    const float* conv_w  = (const float*)d_in[5];
    const float* conv_b  = (const float*)d_in[6];
    const float* Bp[12], *Cp[12];
    for (int i = 0; i < 12; ++i) Bp[i] = (const float*)d_in[7 + i];
    for (int i = 0; i < 12; ++i) Cp[i] = (const float*)d_in[19 + i];
    const float* P_ln1_g = (const float*)d_in[31];
    const float* P_ln1_b = (const float*)d_in[32];
    const float* P_W     = (const float*)d_in[33];
    const float* P_b     = (const float*)d_in[34];
    const float* P_ln2_g = (const float*)d_in[35];
    const float* P_ln2_b = (const float*)d_in[36];
    const float* pos_emb = (const float*)d_in[37];
    const float* dt_W1   = (const float*)d_in[38];
    const float* dt_b1   = (const float*)d_in[39];
    const float* dt_W2   = (const float*)d_in[40];
    const float* dt_b2   = (const float*)d_in[41];
    const float* dt_ln_g = (const float*)d_in[42];
    const float* dt_ln_b = (const float*)d_in[43];
    const float* A_logs  = (const float*)d_in[44];
    const float* Ds      = (const float*)d_in[45];
    const float* gate_W  = (const float*)d_in[46];
    const float* onorm_g = (const float*)d_in[47];
    const float* onorm_b = (const float*)d_in[48];
    const float* out_W   = (const float*)d_in[49];
    const float* out_b   = (const float*)d_in[50];
    const float* op_W1   = (const float*)d_in[51];
    const float* op_W2   = (const float*)d_in[52];
    const float* op_ln_g = (const float*)d_in[53];
    const float* op_ln_b = (const float*)d_in[54];

    // ---- workspace arena -------------------------------------------------
    char* wsb = (char*)d_ws;
    size_t off = 0;
    auto alloc = [&](size_t bytes) -> void* {
        void* p = wsb + off;
        off += (bytes + 255) & ~(size_t)255;
        return p;
    };
    const size_t TOK = 65536;

    _Float16* Wf_in1 = (_Float16*)alloc(32768 * 2);
    _Float16* Wf_in2 = (_Float16*)alloc(16384 * 2);
    _Float16* Wf_B1  = (_Float16*)alloc(1048576 * 2);
    _Float16* Wf_B2  = (_Float16*)alloc(262144 * 2);
    _Float16* Wf_C1  = (_Float16*)alloc(1048576 * 2);
    _Float16* Wf_C2  = (_Float16*)alloc(262144 * 2);
    _Float16* Wf_P   = (_Float16*)alloc(1048576 * 2);
    _Float16* Wf_out = (_Float16*)alloc(262144 * 2);
    _Float16* Wf_op1 = (_Float16*)alloc(32768 * 2);
    _Float16* Wf_op2 = (_Float16*)alloc(131072 * 2);

    float* t1  = (float*)alloc(8388608 * 4);   // region1 start
    float* xfb = (float*)alloc(8388608 * 4);
    float* Abr = (float*)alloc(8388608 * 4);
    float* Apt = (float*)alloc(8388608 * 4);
    float* xz   = (float*)alloc(8388608 * 4);  // region2 start
    float* xh   = (float*)alloc(4194304 * 4);
    float* ybig = (float*)alloc(4194304 * 4);
    float* st_br = (float*)alloc(2048 * 4);
    float* st_pt = (float*)alloc(2048 * 4);
    float* st2   = (float*)alloc(128 * 4);
    float* tB    = (float*)alloc(131072 * 4);
    float* tBln  = (float*)alloc(131072 * 4);
    float* A2    = (float*)alloc(131072 * 4);
    float* Bpre  = (float*)alloc(8192 * 4);
    float* Bsf   = (float*)alloc(8192 * 4);
    float* Cpre  = (float*)alloc(8192 * 4);
    float* Csf   = (float*)alloc(8192 * 4);
    float* tP    = (float*)alloc(131072 * 4);
    float* tPln  = (float*)alloc(131072 * 4);
    float* tdout = (float*)alloc(131072 * 4);
    float* oy    = (float*)alloc(131072 * 4);
    float* gts   = (float*)alloc(32 * 4);
    float* yc    = (float*)alloc(65536 * 4);
    float* ycln  = (float*)alloc(65536 * 4);
    float* yA    = (float*)alloc(65536 * 4);
    float* yin   = (float*)alloc(4194304 * 4);
    float* h1 = t1;    // TOK*512 f32 == region1 exactly (dead by then)
    float* o2 = xz;    // TOK*256 f32 == region2 exactly (dead by then)

    auto cvt = [&](const float* src, _Float16* dst, unsigned K, unsigned N) {
        cvt_tr_kernel<<<dim3((K * N + 255) / 256), 256, 0, stream>>>(src, dst, K, N);
    };
    auto gemm = [&](const float* A, const _Float16* Wt, const float* bias,
                    const float* stats, const float* g, const float* b,
                    float* C, int M, int N, int Kd, int dogelu) {
        dim3 gr(N / 64, M / 64);
        if (stats)        gemm_f16_kernel<1,1,0><<<gr, 256, 0, stream>>>(A, Wt, bias, stats, g, b, C, M, N, Kd);
        else if (dogelu)  gemm_f16_kernel<0,0,1><<<gr, 256, 0, stream>>>(A, Wt, bias, stats, g, b, C, M, N, Kd);
        else if (bias)    gemm_f16_kernel<0,1,0><<<gr, 256, 0, stream>>>(A, Wt, bias, stats, g, b, C, M, N, Kd);
        else              gemm_f16_kernel<0,0,0><<<gr, 256, 0, stream>>>(A, Wt, bias, stats, g, b, C, M, N, Kd);
    };
    auto ln = [&](const float* X, const float* g, const float* b, float* Y,
                  int M, int C) {
        ln_kernel<<<dim3((M + 3) / 4), 128, 0, stream>>>(X, g, b, Y, M, C);
    };
    auto rstats = [&](const float* X, float* st, int M, int C) {
        row_stats_kernel<<<dim3((M + 3) / 4), 128, 0, stream>>>(X, st, M, C);
    };
    auto ew = [&](unsigned n) { return dim3((n + 255) / 256); };

    // ---- weight conversion (f32 [K][N] -> f16 [N][K]) --------------------
    cvt(in_W1, Wf_in1, 256, 128);   cvt(in_W2, Wf_in2, 128, 128);
    cvt(Bp[2], Wf_B1, 8192, 128);   cvt(Bp[8], Wf_B2, 2048, 128);
    cvt(Cp[2], Wf_C1, 8192, 128);   cvt(Cp[8], Wf_C2, 2048, 128);
    cvt(P_W,   Wf_P,  8192, 128);
    cvt(out_W, Wf_out, 64, 4096);
    cvt(op_W1, Wf_op1, 64, 512);    cvt(op_W2, Wf_op2, 512, 256);

    // ---- in-projection ---------------------------------------------------
    gemm(x,  Wf_in1, nullptr, nullptr, nullptr, nullptr, t1, TOK, 128, 256, 1);
    gemm(t1, Wf_in2, nullptr, nullptr, nullptr, nullptr, xz, TOK, 128, 128, 0);
    ln(xz, in_ln_g, in_ln_b, xz, TOK, 128);

    // ---- conv + x_fb -----------------------------------------------------
    conv_silu_kernel<<<ew(4194304), 256, 0, stream>>>(xz, conv_w, conv_b, xh);
    xfb_kernel<<<ew(8388608), 256, 0, stream>>>(xh, xfb);

    // ---- B/C branches (shared gather, LN folded into GEMM A-load) -------
    gather_branch_kernel<<<ew(8388608), 256, 0, stream>>>(xfb, Abr);
    rstats(Abr, st_br, 1024, 8192);
    gemm(Abr, Wf_B1, Bp[3], st_br, Bp[0], Bp[1], tB, 1024, 128, 8192, 0);
    ln(tB, Bp[4], Bp[5], tBln, 1024, 128);
    gather_stage2_kernel<<<ew(131072), 256, 0, stream>>>(tBln, A2);
    rstats(A2, st2, 64, 2048);
    gemm(A2, Wf_B2, Bp[9], st2, Bp[6], Bp[7], Bpre, 64, 128, 2048, 0);
    ln(Bpre, Bp[10], Bp[11], Bsf, 64, 128);
    gemm(Abr, Wf_C1, Cp[3], st_br, Cp[0], Cp[1], tB, 1024, 128, 8192, 0);
    ln(tB, Cp[4], Cp[5], tBln, 1024, 128);
    gather_stage2_kernel<<<ew(131072), 256, 0, stream>>>(tBln, A2);
    rstats(A2, st2, 64, 2048);
    gemm(A2, Wf_C2, Cp[9], st2, Cp[6], Cp[7], Cpre, 64, 128, 2048, 0);
    ln(Cpre, Cp[10], Cp[11], Csf, 64, 128);

    // ---- P path ----------------------------------------------------------
    gather_patch_kernel<<<ew(8388608), 256, 0, stream>>>(xfb, Apt);
    rstats(Apt, st_pt, 1024, 8192);
    gemm(Apt, Wf_P, P_b, st_pt, P_ln1_g, P_ln1_b, tP, 1024, 128, 8192, 0);
    ln(tP, P_ln2_g, P_ln2_b, tPln, 1024, 128);
    posadd_kernel<<<ew(131072), 256, 0, stream>>>(tPln, pos_emb);

    // ---- dt MLP + selective scan ----------------------------------------
    dt_kernel<<<dim3(16), 128, 0, stream>>>(tPln, dt_W1, dt_b1, dt_W2, dt_b2,
                                            dt_ln_g, dt_ln_b, tdout);
    scan_kernel<<<dim3(16), 128, 0, stream>>>(tPln, tdout, Bsf, Csf,
                                              A_logs, Ds, oy);

    // ---- gate / combine / onorm -----------------------------------------
    gate_kernel<<<dim3(1), 16, 0, stream>>>(oy, gate_W, gts);
    combine_kernel<<<ew(65536), 256, 0, stream>>>(oy, gts, yc);
    ln(yc, onorm_g, onorm_b, ycln, 1024, 64);
    rearr_y_kernel<<<ew(65536), 256, 0, stream>>>(ycln, yA);

    // ---- out projection + gating by z -----------------------------------
    gemm(yA, Wf_out, out_b, nullptr, nullptr, nullptr, ybig, 1024, 4096, 64, 0);
    mulz_kernel<<<ew(4194304), 256, 0, stream>>>(ybig, xz, yin);

    // ---- output MLP (h1/o2 alias dead regions) ---------------------------
    gemm(yin, Wf_op1, nullptr, nullptr, nullptr, nullptr, h1, TOK, 512, 64, 1);
    gemm(h1,  Wf_op2, nullptr, nullptr, nullptr, nullptr, o2, TOK, 256, 512, 0);
    ln(o2, op_ln_g, op_ln_b, (float*)d_out, TOK, 256);
}
